// PLANAttention_44959717654697
// MI455X (gfx1250) — matrix operand order repaired and checked by reading, not compiled, not run on hardware
//
#include <hip/hip_runtime.h>

// ---------------------------------------------------------------------------
// PLAN attention for MI455X (gfx1250, wave32, WMMA)
// ---------------------------------------------------------------------------

typedef __attribute__((ext_vector_type(16))) __bf16 v16bf;
typedef __attribute__((ext_vector_type(8)))  float  v8f;

#define OUTCH 256
#define HWT   4096     // 64*64 top
#define HWB   16384    // 128*128 bot

// ---------------------------------------------------------------------------
// Generic 256(out) x M x 256(K) GEMM: Y[b,o,m] = sum_c W[o,c] * X[b,c,m]
// W: (256,256) f32 row-major.  X,Y: (B,256,HW) f32.  bf16 WMMA, f32 accum.
// Block tile 64(O) x 128(M), K-chunks of 32, double-buffered LDS, software
// pipelined global->reg->LDS staging (batched loads, packed b64 LDS stores).
// 256 threads = 8 waves; each wave owns a 16x64 strip -> 4 WMMAs per chunk.
// LDS fragment layouts give contiguous 2x ds_load_b128 per lane:
//   Wt [o][k]  pitch 40 bf16 (80B: 16B-aligned, conflict-free b128 reads)
//   XtT[m][k]  pitch 40 bf16 (K contiguous per lane)
// ---------------------------------------------------------------------------
__global__ __launch_bounds__(256) void gemm_wmma_kernel(
    const float* __restrict__ Wm,
    const float* __restrict__ X,
    float* __restrict__ Y,
    int HW)
{
    __shared__ __bf16 Wt [2][64][40];    // double buffered
    __shared__ __bf16 XtT[2][128][40];

    const int tid   = threadIdx.x;
    const int lane  = tid & 31;
    const int wid   = tid >> 5;
    const int lhalf = lane >> 4;      // 0|1
    const int lmod  = lane & 15;

    const int m0  = blockIdx.x * 128;         // spatial index across B*HW
    const int o0  = blockIdx.y * 64;
    const int b   = m0 / HW;
    const int hw0 = m0 - b * HW;

    const int oLocal = (wid >> 1) << 4;       // 0,16,32,48
    const int mGroup = (wid & 1) << 6;        // 0 | 64

    const size_t xplane = (size_t)b * OUTCH * HW;

    // ---- staging registers ----
    float wr[8];    // W: 2 float4 groups (row r, cols c4..c4+3)
    float xr[16];   // X: 4 groups of 4 consecutive k for one column

    // per-thread staging coordinates (constant across chunks)
    const int wg0_r  = tid >> 3,          wg0_c = (tid & 7) << 2;
    const int wg1_r  = (256 + tid) >> 3,  wg1_c = ((256 + tid) & 7) << 2;

    auto load_stage = [&](int kk) {
        const float4 v0 = *(const float4*)&Wm[(o0 + wg0_r) * OUTCH + kk + wg0_c];
        const float4 v1 = *(const float4*)&Wm[(o0 + wg1_r) * OUTCH + kk + wg1_c];
        wr[0] = v0.x; wr[1] = v0.y; wr[2] = v0.z; wr[3] = v0.w;
        wr[4] = v1.x; wr[5] = v1.y; wr[6] = v1.z; wr[7] = v1.w;
        #pragma unroll
        for (int e = 0; e < 4; ++e) {
            int g   = e * 256 + tid;          // 0..1023
            int col = g & 127;
            int kb  = (g >> 7) << 2;          // 0,4,...,28
            #pragma unroll
            for (int j = 0; j < 4; ++j)
                xr[e * 4 + j] = X[xplane + (size_t)(kk + kb + j) * HW + hw0 + col];
        }
    };

    auto store_stage = [&](int buf) {
        union Pack4 { __bf16 h[4]; unsigned long long u; };
        {
            Pack4 p;
            p.h[0] = (__bf16)wr[0]; p.h[1] = (__bf16)wr[1];
            p.h[2] = (__bf16)wr[2]; p.h[3] = (__bf16)wr[3];
            *(unsigned long long*)&Wt[buf][wg0_r][wg0_c] = p.u;
            p.h[0] = (__bf16)wr[4]; p.h[1] = (__bf16)wr[5];
            p.h[2] = (__bf16)wr[6]; p.h[3] = (__bf16)wr[7];
            *(unsigned long long*)&Wt[buf][wg1_r][wg1_c] = p.u;
        }
        #pragma unroll
        for (int e = 0; e < 4; ++e) {
            int g   = e * 256 + tid;
            int col = g & 127;
            int kb  = (g >> 7) << 2;
            Pack4 p;
            p.h[0] = (__bf16)xr[e * 4 + 0]; p.h[1] = (__bf16)xr[e * 4 + 1];
            p.h[2] = (__bf16)xr[e * 4 + 2]; p.h[3] = (__bf16)xr[e * 4 + 3];
            *(unsigned long long*)&XtT[buf][col][kb] = p.u;
        }
    };

    v8f acc[4];
    #pragma unroll
    for (int t = 0; t < 4; ++t)
        acc[t] = (v8f){0.f,0.f,0.f,0.f,0.f,0.f,0.f,0.f};

    // ---- software-pipelined main loop: one barrier per K-chunk ----
    load_stage(0);
    int buf = 0;
    for (int kk = 0; kk < OUTCH; kk += 32) {
        store_stage(buf);                       // waits on in-flight loads
        if (kk + 32 < OUTCH)
            load_stage(kk + 32);                // issue next chunk's loads
        __syncthreads();

        // A fragment (16x32 bf16, contiguous k = i + 16*lhalf): two b128 per lane
        v16bf a;
        #pragma unroll
        for (int i = 0; i < 16; ++i) {
            int k = i + (lhalf << 4);
            a[i] = Wt[buf][oLocal + lmod][k];
        }
        // B fragments (32x16): lane col = tile + lmod, k = i + 16*lhalf
        v16bf bf[4];
        #pragma unroll
        for (int t = 0; t < 4; ++t) {
            int col = mGroup + (t << 4) + lmod;
            #pragma unroll
            for (int i = 0; i < 16; ++i)
                bf[t][i] = XtT[buf][col][i + (lhalf << 4)];
        }

        #pragma unroll
        for (int t = 0; t < 4; ++t)
            acc[t] = __builtin_amdgcn_wmma_f32_16x16x32_bf16(
                         false, a, false, bf[t], (short)0, acc[t], false, false);
        buf ^= 1;
    }

    // ---- store D: VGPR r -> row m = r + 8*lhalf ; col n = lane&15 ----
    const size_t yplane = (size_t)b * OUTCH * HW;
    #pragma unroll
    for (int t = 0; t < 4; ++t) {
        #pragma unroll
        for (int r = 0; r < 8; ++r) {
            int orow = o0 + oLocal + r + (lhalf << 3);
            Y[yplane + (size_t)orow * HW + hw0 + mGroup + (t << 4) + lmod] = acc[t][r];
        }
    }
}

// ---------------------------------------------------------------------------
// Attention core. One thread per (b, head, top pixel).
// K/Q/V rows: row0 = top proj at pixel; rows 1..9 = 3x3 window (stride 2,
// pad 1) of bot proj. weight[k][q] = <K_k,Q_q>/sqrt(32);
// out[q] = sum_k weight[k][q] * V_k.
// q==0 -> top_ret (B,256,HWT);  q>0 -> bot_pat (B,8,9,32,HWT)
// ---------------------------------------------------------------------------
__global__ __launch_bounds__(256) void attn_kernel(
    const float* __restrict__ tK, const float* __restrict__ tQ,
    const float* __restrict__ tV,
    const float* __restrict__ bK, const float* __restrict__ bQ,
    const float* __restrict__ bV,
    float* __restrict__ top_ret, float* __restrict__ bot_pat,
    int total)
{
    int gid = blockIdx.x * 256 + threadIdx.x;
    if (gid >= total) return;

    const int pos  = gid & 4095;
    const int head = (gid >> 12) & 7;
    const int b    = gid >> 15;
    const int h    = pos >> 6;
    const int w    = pos & 63;

    const float inv = 0.17677669529663687f;   // 1/sqrt(32)

    const size_t tbase = ((size_t)(b * OUTCH + head * 32)) * HWT + pos;
    const size_t bbase = ((size_t)(b * OUTCH + head * 32)) * HWB;

    #pragma unroll 1
    for (int q = 0; q < 10; ++q) {
        // ---- load query row q (zeros if padded-out) ----
        float Qr[32];
        #pragma unroll
        for (int d = 0; d < 32; ++d) Qr[d] = 0.f;
        if (q == 0) {
            #pragma unroll
            for (int d = 0; d < 32; ++d) Qr[d] = tQ[tbase + (size_t)d * HWT];
        } else {
            int n = q - 1, i = n / 3, j = n - i * 3;
            int yy = 2 * h + i - 1, xx = 2 * w + j - 1;
            if ((unsigned)yy < 128u && (unsigned)xx < 128u) {
                size_t o = bbase + (size_t)(yy * 128 + xx);
                #pragma unroll
                for (int d = 0; d < 32; ++d) Qr[d] = bQ[o + (size_t)d * HWB];
            }
        }

        float acc[32];
        #pragma unroll
        for (int d = 0; d < 32; ++d) acc[d] = 0.f;

        // ---- accumulate over the 10 key/value rows ----
        #pragma unroll 1
        for (int k = 0; k < 10; ++k) {
            float s = 0.f;
            if (k == 0) {
                #pragma unroll
                for (int d = 0; d < 32; ++d)
                    s += tK[tbase + (size_t)d * HWT] * Qr[d];
                s *= inv;
                #pragma unroll
                for (int d = 0; d < 32; ++d)
                    acc[d] += s * tV[tbase + (size_t)d * HWT];
            } else {
                int n = k - 1, i = n / 3, j = n - i * 3;
                int yy = 2 * h + i - 1, xx = 2 * w + j - 1;
                if ((unsigned)yy < 128u && (unsigned)xx < 128u) {
                    size_t o = bbase + (size_t)(yy * 128 + xx);
                    #pragma unroll
                    for (int d = 0; d < 32; ++d)
                        s += bK[o + (size_t)d * HWB] * Qr[d];
                    s *= inv;
                    #pragma unroll
                    for (int d = 0; d < 32; ++d)
                        acc[d] += s * bV[o + (size_t)d * HWB];
                }
            }
        }

        // ---- store ----
        if (q == 0) {
            #pragma unroll
            for (int d = 0; d < 32; ++d)
                top_ret[tbase + (size_t)d * HWT] = acc[d];
        } else {
            size_t obase = ((((size_t)b * 8 + head) * 9 + (q - 1)) * 32) * HWT + pos;
            #pragma unroll
            for (int d = 0; d < 32; ++d)
                bot_pat[obase + (size_t)d * HWT] = acc[d];
        }
    }
}

// ---------------------------------------------------------------------------
// Gather-style fold (overlap-add inverse of stride-2 3x3 unfold, pad 1).
// bot_ret[b, head*32+v, y, x] = sum over windows (i,j,h,w) with
//   i + 2h == y+1, j + 2w == x+1, of bot_pat[b,head,i*3+j,v,h*64+w]
// ---------------------------------------------------------------------------
__global__ __launch_bounds__(256) void fold_kernel(
    const float* __restrict__ bot_pat,
    float* __restrict__ bot_ret,
    int total)
{
    int gid = blockIdx.x * 256 + threadIdx.x;
    if (gid >= total) return;

    const int x  = gid & 127;
    const int y  = (gid >> 7) & 127;
    const int ch = (gid >> 14) & 255;
    const int b  = gid >> 22;
    const int head = ch >> 5;
    const int v    = ch & 31;

    float s = 0.f;
    #pragma unroll
    for (int i = 0; i < 3; ++i) {
        int hy = y + 1 - i;
        if (hy < 0 || (hy & 1)) continue;
        int hh = hy >> 1;
        if (hh >= 64) continue;
        #pragma unroll
        for (int j = 0; j < 3; ++j) {
            int wx = x + 1 - j;
            if (wx < 0 || (wx & 1)) continue;
            int ww = wx >> 1;
            if (ww >= 64) continue;
            s += bot_pat[((((size_t)b * 8 + head) * 9 + (i * 3 + j)) * 32 + v) * HWT
                         + hh * 64 + ww];
        }
    }
    bot_ret[(size_t)gid] = s;
}

// ---------------------------------------------------------------------------
// Launch
// ---------------------------------------------------------------------------
extern "C" void kernel_launch(void* const* d_in, const int* in_sizes, int n_in,
                              void* d_out, int out_size, void* d_ws, size_t ws_size,
                              hipStream_t stream)
{
    const float* top_feat   = (const float*)d_in[0];
    const float* bot_feat   = (const float*)d_in[1];
    const float* W_top_key  = (const float*)d_in[2];
    const float* W_top_qry  = (const float*)d_in[3];
    const float* W_top_val  = (const float*)d_in[4];
    const float* W_bot_key  = (const float*)d_in[5];
    const float* W_bot_qry  = (const float*)d_in[6];
    const float* W_bot_val  = (const float*)d_in[7];
    const float* W_top_fuse = (const float*)d_in[8];
    const float* W_bot_fuse = (const float*)d_in[9];
    float* out = (float*)d_out;

    const int B = in_sizes[0] / (OUTCH * HWT);   // = 2

    // workspace layout (f32)
    float* ws = (float*)d_ws;
    size_t off = 0;
    float* tK = ws + off; off += (size_t)B * OUTCH * HWT;
    float* tQ = ws + off; off += (size_t)B * OUTCH * HWT;
    float* tV = ws + off; off += (size_t)B * OUTCH * HWT;
    float* bK = ws + off; off += (size_t)B * OUTCH * HWB;
    float* bQ = ws + off; off += (size_t)B * OUTCH * HWB;
    float* bV = ws + off; off += (size_t)B * OUTCH * HWB;
    float* top_ret = ws + off; off += (size_t)B * OUTCH * HWT;
    float* bot_pat = ws + off; off += (size_t)B * 8 * 9 * 32 * HWT;
    float* bot_ret = ws + off; off += (size_t)B * OUTCH * HWB;
    (void)ws_size; (void)n_in; (void)out_size;

    dim3 blk(256);
    dim3 gT(B * HWT / 128, OUTCH / 64);   // top-resolution GEMM grid
    dim3 gB(B * HWB / 128, OUTCH / 64);   // bot-resolution GEMM grid

    // projections (1x1 convs) -> WMMA GEMMs
    gemm_wmma_kernel<<<gT, blk, 0, stream>>>(W_top_key, top_feat, tK, HWT);
    gemm_wmma_kernel<<<gT, blk, 0, stream>>>(W_top_qry, top_feat, tQ, HWT);
    gemm_wmma_kernel<<<gT, blk, 0, stream>>>(W_top_val, top_feat, tV, HWT);
    gemm_wmma_kernel<<<gB, blk, 0, stream>>>(W_bot_key, bot_feat, bK, HWB);
    gemm_wmma_kernel<<<gB, blk, 0, stream>>>(W_bot_qry, bot_feat, bQ, HWB);
    gemm_wmma_kernel<<<gB, blk, 0, stream>>>(W_bot_val, bot_feat, bV, HWB);

    // attention core
    int attn_total = B * 8 * HWT;
    attn_kernel<<<(attn_total + 255) / 256, blk, 0, stream>>>(
        tK, tQ, tV, bK, bQ, bV, top_ret, bot_pat, attn_total);

    // fold (overlap-add) back to bot resolution
    int fold_total = B * OUTCH * HWB;
    fold_kernel<<<(fold_total + 255) / 256, blk, 0, stream>>>(
        bot_pat, bot_ret, fold_total);

    // fuse 1x1 convs -> outputs (top first, then bot, flat-concatenated)
    gemm_wmma_kernel<<<gT, blk, 0, stream>>>(W_top_fuse, top_ret, out, HWT);
    gemm_wmma_kernel<<<gB, blk, 0, stream>>>(W_bot_fuse, bot_ret,
                                             out + (size_t)B * OUTCH * HWT, HWB);
}
